// ScaleDotProductAttention_83485574300204
// MI455X (gfx1250) — matrix operand order repaired and checked by reading, not compiled, not run on hardware
//
#include <hip/hip_runtime.h>
#include <math.h>

// ---------------------------------------------------------------------------
// Fused ScaledDotProductAttention for MI455X (gfx1250, wave32, WMMA).
//   q = X @ Wq^T + bq ; k,v likewise                 (bf16 out, fp32-accum WMMA)
//   S = (q @ k^T) / sqrt(D) -> masked softmax        (bf16 probs, fp32 math)
//   O = P @ v                                        (fp32 out)
// bf16 WMMA (16x16x32) chosen over f32 WMMA (16x16x4): 8x K per issue, and the
// problem is compute-bound (~120 GFLOP vs ~350MB traffic -> ~15us at 23.3TB/s).
// bf16 staging uses GLOBAL_LOAD_ASYNC_TO_LDS_B128 (ASYNCcnt) on the CDNA5
// async path; fp32 inputs convert in-register (v_cvt_pk_bf16_f32) -> ds_store.
// ---------------------------------------------------------------------------

typedef __bf16 bf16_t;
typedef __bf16 v16bf __attribute__((ext_vector_type(16)));
typedef float  v8f   __attribute__((ext_vector_type(8)));

#define BLK_M 64
#define BLK_N 128
#define BLK_K 32
#define LDS_PAD 8   // row stride 40 bf16 = 80B (16B aligned)

static __device__ __forceinline__ bf16_t f2bf(float f) { return (bf16_t)f; }

// LDS byte offset of a generic pointer into shared memory (AS3 ptrtoint).
static __device__ __forceinline__ unsigned lds_addr(const void* p) {
    typedef __attribute__((address_space(3))) const void as3_void;
    return (unsigned)(unsigned long long)(as3_void*)p;
}

// Per-lane async copy: 16 bytes global -> LDS, tracked by ASYNCcnt.
static __device__ __forceinline__ void async_ld_b128(unsigned lds, const void* g) {
    asm volatile("global_load_async_to_lds_b128 %0, %1, off"
                 :: "v"(lds), "v"((unsigned long long)(size_t)g)
                 : "memory");
}
static __device__ __forceinline__ void wait_async0() {
    asm volatile("s_wait_asynccnt 0x0" ::: "memory");
}

// ---------------------------------------------------------------------------
// Tiled WMMA GEMM:  C[M,N] = A[M,K] @ op(B) (+ bias)
//   BTRANS=true : B row-major [N,K] (computes A @ B^T)
//   BTRANS=false: B row-major [K,N]
// Block: 256 threads = 8 waves, 64x128 output tile, K stepped by 32.
// Wave w: rows (w&3)*16, cols (w>>2)*64 -> four 16x16 accumulators sharing
// one A fragment (4 WMMAs per staged A-frag). grid.z = batch.
// ---------------------------------------------------------------------------
template <typename AT, typename BT, typename OT, bool BTRANS, bool BIAS>
__global__ __launch_bounds__(256)
void gemm_wmma_bf16(const AT* __restrict__ A, const BT* __restrict__ Bm,
                    const float* __restrict__ bias, OT* __restrict__ C,
                    int M, int N, int K,
                    long long sA, long long sB, long long sC) {
    __shared__ __align__(16) bf16_t As[BLK_M][BLK_K + LDS_PAD];
    __shared__ __align__(16) bf16_t Bs[BLK_N][BLK_K + LDS_PAD];  // Bs[n][k]

    constexpr bool ASYNC_A = (sizeof(AT) == 2);
    constexpr bool ASYNC_B = (BTRANS && sizeof(BT) == 2);

    const int tid  = threadIdx.x;
    const int lane = tid & 31;
    const int wave = tid >> 5;
    const int wm   = (wave & 3) * 16;   // row offset inside 64x128 tile
    const int wn   = (wave >> 2) * 64;  // col offset inside 64x128 tile

    const int m0 = blockIdx.y * BLK_M;
    const int n0 = blockIdx.x * BLK_N;

    const AT* Ab = A  + (size_t)blockIdx.z * (size_t)sA;
    const BT* Bb = Bm + (size_t)blockIdx.z * (size_t)sB;
    OT*       Cb = C  + (size_t)blockIdx.z * (size_t)sC;

    v8f acc[4] = {v8f{}, v8f{}, v8f{}, v8f{}};

    for (int k0 = 0; k0 < K; k0 += BLK_K) {
        // ---- A tile: 64 x 32
        if constexpr (!ASYNC_A) {
            // fp32 -> bf16 convert while staging
            #pragma unroll
            for (int i = 0; i < 2; ++i) {
                int idx = tid + i * 256;        // 512 float4s
                int r = idx >> 3;
                int c = (idx & 7) * 4;
                const float4 vv = *(const float4*)(Ab + (size_t)(m0 + r) * K + k0 + c);
                As[r][c + 0] = f2bf(vv.x); As[r][c + 1] = f2bf(vv.y);
                As[r][c + 2] = f2bf(vv.z); As[r][c + 3] = f2bf(vv.w);
            }
        } else {
            int r = tid >> 2;
            int c = (tid & 3) * 8;              // 8 bf16 = 16B per lane
            async_ld_b128(lds_addr(&As[r][c]), Ab + (size_t)(m0 + r) * K + k0 + c);
        }

        // ---- B tile into Bs[n][k] (128 x 32)
        if constexpr (BTRANS) {
            if constexpr (!ASYNC_B) {
                #pragma unroll
                for (int i = 0; i < 4; ++i) {
                    int idx = tid + i * 256;    // 1024 float4s
                    int r = idx >> 3;           // 0..127
                    int c = (idx & 7) * 4;
                    const float4 vv = *(const float4*)(Bb + (size_t)(n0 + r) * K + k0 + c);
                    Bs[r][c + 0] = f2bf(vv.x); Bs[r][c + 1] = f2bf(vv.y);
                    Bs[r][c + 2] = f2bf(vv.z); Bs[r][c + 3] = f2bf(vv.w);
                }
            } else {
                #pragma unroll
                for (int i = 0; i < 2; ++i) {
                    int idx = tid + i * 256;    // 512 b128s
                    int r = idx >> 2;           // 0..127
                    int c = (idx & 3) * 8;
                    async_ld_b128(lds_addr(&Bs[r][c]), Bb + (size_t)(n0 + r) * K + k0 + c);
                }
            }
        } else {
            // B row-major [K,N] (bf16): transpose into Bs[n][k] while staging
            #pragma unroll
            for (int i = 0; i < 2; ++i) {
                int idx = tid + i * 256;        // 512 x (8 bf16)
                int k = idx >> 4;               // 0..31
                int n = (idx & 15) * 8;         // 0..120
                bf16_t tmp[8];
                *(uint4*)tmp = *(const uint4*)(Bb + (size_t)(k0 + k) * N + n0 + n);
                #pragma unroll
                for (int j = 0; j < 8; ++j) Bs[n + j][k] = tmp[j];
            }
        }

        if constexpr (ASYNC_A || ASYNC_B) wait_async0();
        __syncthreads();

        // ---- A fragment (16x32 bf16, ISA 7.12.2 layout)
        const int mrow = wm + (lane & 15);
        const int kb   = (lane >> 4) * 8;
        v16bf af;
        #pragma unroll
        for (int j = 0; j < 4; ++j) {
            af[2 * j]         = As[mrow][kb + 2 * j];
            af[2 * j + 1]     = As[mrow][kb + 2 * j + 1];
            af[8 + 2 * j]     = As[mrow][16 + kb + 2 * j];
            af[8 + 2 * j + 1] = As[mrow][16 + kb + 2 * j + 1];
        }

        // ---- four B fragments, one WMMA each, sharing the A fragment
        const int ncol = lane & 15;
        const int ks   = (lane >> 4) * 8;
        #pragma unroll
        for (int t = 0; t < 4; ++t) {
            v16bf bf;
            #pragma unroll
            for (int s = 0; s < 16; ++s) bf[s] = Bs[wn + t * 16 + ncol][ks + (s & 7) + ((s >> 3) << 4)];
            acc[t] = __builtin_amdgcn_wmma_f32_16x16x32_bf16(false, af, false, bf,
                                                             (short)0, acc[t],
                                                             false, false);
        }
        __syncthreads();
    }

    // ---- epilogue: C/D layout = 8 VGPRs; lanes 0-15 -> M=r, 16-31 -> M=r+8
    const int mo = m0 + wm + (lane >> 4) * 8;
    const int nc = lane & 15;
    #pragma unroll
    for (int t = 0; t < 4; ++t) {
        const int n = n0 + wn + t * 16 + nc;
        float bv = 0.0f;
        if constexpr (BIAS) bv = bias[n];
        #pragma unroll
        for (int r = 0; r < 8; ++r) {
            float v = acc[t][r] + bv;
            Cb[(size_t)(mo + r) * N + n] = (OT)v;
        }
    }
}

// ---------------------------------------------------------------------------
// Row softmax with scale + causal + padding masks, in-place on bf16 scores.
// grid = (S, B), block = 256 (8 waves). Row length S = 2048 -> 8 elems/thread.
// ---------------------------------------------------------------------------
__global__ __launch_bounds__(256)
void softmax_mask_kernel(bf16_t* __restrict__ sc,
                         const unsigned char* __restrict__ pad,
                         int S, float scale) {
    const int q = blockIdx.x;
    const int b = blockIdx.y;
    const int tid = threadIdx.x;
    const size_t base = ((size_t)b * S + q) * (size_t)S;

    float xv[8];
    float mx = -INFINITY;
    #pragma unroll
    for (int i = 0; i < 8; ++i) {
        const int k = tid + i * 256;
        float x = (float)sc[base + k] * scale;
        const bool masked = (k > q) || (pad[(size_t)b * S + k] != 0);
        x = masked ? -INFINITY : x;
        xv[i] = x;
        mx = fmaxf(mx, x);
    }
    #pragma unroll
    for (int off = 16; off > 0; off >>= 1)
        mx = fmaxf(mx, __shfl_xor(mx, off, 32));

    __shared__ float redm[8];
    __shared__ float reds[8];
    if ((tid & 31) == 0) redm[tid >> 5] = mx;
    __syncthreads();
    float bm = redm[0];
    #pragma unroll
    for (int w = 1; w < 8; ++w) bm = fmaxf(bm, redm[w]);

    float pv[8];
    float sum = 0.0f;
    #pragma unroll
    for (int i = 0; i < 8; ++i) {
        const float p = __expf(xv[i] - bm);   // exp(-inf) -> 0 for masked
        pv[i] = p;
        sum += p;
    }
    #pragma unroll
    for (int off = 16; off > 0; off >>= 1)
        sum += __shfl_xor(sum, off, 32);
    if ((tid & 31) == 0) reds[tid >> 5] = sum;
    __syncthreads();
    float bs = 0.0f;
    #pragma unroll
    for (int w = 0; w < 8; ++w) bs += reds[w];
    const float inv = 1.0f / bs;

    #pragma unroll
    for (int i = 0; i < 8; ++i) {
        const int k = tid + i * 256;
        sc[base + k] = f2bf(pv[i] * inv);
    }
}

// ---------------------------------------------------------------------------
extern "C" void kernel_launch(void* const* d_in, const int* in_sizes, int n_in,
                              void* d_out, int out_size, void* d_ws, size_t ws_size,
                              hipStream_t stream) {
    (void)in_sizes; (void)n_in; (void)out_size; (void)ws_size;

    constexpr int B = 4, S = 2048, E = 1024, D = 1024;
    constexpr int M = B * S;  // 8192

    const float* query = (const float*)d_in[0];
    const float* key   = (const float*)d_in[1];
    const float* value = (const float*)d_in[2];
    const float* Wq    = (const float*)d_in[3];
    const float* bq    = (const float*)d_in[4];
    const float* Wk    = (const float*)d_in[5];
    const float* bk    = (const float*)d_in[6];
    const float* Wv    = (const float*)d_in[7];
    const float* bv    = (const float*)d_in[8];
    const unsigned char* pad = (const unsigned char*)d_in[9];  // bool [B,1,S]
    // d_in[10] = look_ahead_mask (scalar 1): causal mask applied unconditionally.

    float* out = (float*)d_out;
    char*  ws  = (char*)d_ws;

    bf16_t* qb = (bf16_t*)(ws);                            // 16 MB
    bf16_t* kb = (bf16_t*)(ws + ((size_t)16 << 20));       // 16 MB
    bf16_t* vb = (bf16_t*)(ws + ((size_t)32 << 20));       // 16 MB
    bf16_t* sc = (bf16_t*)(ws + ((size_t)48 << 20));       // B*S*S bf16 = 32 MB

    const dim3 blk(256);

    // 1) QKV projections: [M,E] @ W^T[E,D] + b -> bf16
    {
        const dim3 grd(D / BLK_N, M / BLK_M, 1);
        gemm_wmma_bf16<float, float, bf16_t, true, true>
            <<<grd, blk, 0, stream>>>(query, Wq, bq, qb, M, D, E, 0, 0, 0);
        gemm_wmma_bf16<float, float, bf16_t, true, true>
            <<<grd, blk, 0, stream>>>(key, Wk, bk, kb, M, D, E, 0, 0, 0);
        gemm_wmma_bf16<float, float, bf16_t, true, true>
            <<<grd, blk, 0, stream>>>(value, Wv, bv, vb, M, D, E, 0, 0, 0);
    }

    // 2) Scores: per batch q[S,D] @ k[S,D]^T -> bf16 [B,S,S] (unscaled)
    {
        const dim3 grd(S / BLK_N, S / BLK_M, B);
        gemm_wmma_bf16<bf16_t, bf16_t, bf16_t, true, false>
            <<<grd, blk, 0, stream>>>(qb, kb, nullptr, sc, S, S, D,
                                      (long long)S * D, (long long)S * D,
                                      (long long)S * S);
    }

    // 3) Masked softmax (scale = 1/sqrt(1024) = 1/32), in-place bf16 probs
    {
        const dim3 grd(S, B);
        softmax_mask_kernel<<<grd, blk, 0, stream>>>(sc, pad, S, 1.0f / 32.0f);
    }

    // 4) Output: per batch P[S,S] @ v[S,D] -> fp32 [B,S,D]
    {
        const dim3 grd(D / BLK_N, S / BLK_M, B);
        gemm_wmma_bf16<bf16_t, bf16_t, float, false, false>
            <<<grd, blk, 0, stream>>>(sc, vb, nullptr, out, S, D, S,
                                      (long long)S * S, (long long)S * D,
                                      (long long)S * D);
    }
}
